// ShiftedWindowAttention_20796231647321
// MI455X (gfx1250) — matrix operand order repaired and checked by reading, not compiled, not run on hardware
//
#include <hip/hip_runtime.h>

#define WSZ 8
#define HEADS 8
#define HC 32
#define CH 256
#define HW 256
#define NWIN 32
#define TOK 64

// per-wave scratch region inside dynamic LDS
#define WAVE_REGION (28 * 1024)
#define LDS_TOTAL (32 * 1024 + 4 * WAVE_REGION)   // 144 KB

typedef __attribute__((ext_vector_type(16))) __bf16 v16bf;
typedef __attribute__((ext_vector_type(8)))  float  v8f;

union Frag { v16bf v; unsigned short u[16]; uint4 q4[2]; };

static __device__ __forceinline__ unsigned short f2bf(float f) {
    unsigned int u = __float_as_uint(f);
    u += 0x7FFFu + ((u >> 16) & 1u);              // round-to-nearest-even
    return (unsigned short)(u >> 16);
}

#if __has_builtin(__builtin_amdgcn_cvt_pk_bf16_f32)
static __device__ __forceinline__ unsigned pack2bf(float a, float b) {
    typedef __attribute__((ext_vector_type(2))) __bf16 v2bf;
    v2bf r = __builtin_amdgcn_cvt_pk_bf16_f32(a, b);
    return __builtin_bit_cast(unsigned, r);
}
#else
static __device__ __forceinline__ unsigned pack2bf(float a, float b) {
    return (unsigned)f2bf(a) | ((unsigned)f2bf(b) << 16);
}
#endif

static __device__ __forceinline__ v8f wmma_bf16(v16bf a, v16bf b, v8f c) {
    return __builtin_amdgcn_wmma_f32_16x16x32_bf16(
        false, a, false, b, (short)0, c, false, false);
}

// A fragment (16x32 bf16, row-major src): lane half 0 -> K{b+0..7, b+16..23},
// lane half 1 -> K{b+8..15, b+24..31}  (ISA 7.12.2 16-bit A layout)
static __device__ __forceinline__ v16bf ld_afrag(const unsigned short* p0, int stride,
                                                 int row, int kbase, int half) {
    Frag f;
    const unsigned short* p = p0 + row * stride + kbase + (half ? 8 : 0);
    f.q4[0] = *(const uint4*)(p);
    f.q4[1] = *(const uint4*)(p + 16);
    return f.v;
}

// B fragment (32x16 bf16): column N = lane&15, lane half 0 -> K{b+0..7, b+16..23}, lane half 1 -> K{b+8..15, b+24..31}
static __device__ __forceinline__ v16bf ld_bfrag(const unsigned short* p0, int stride,
                                                 int col, int kbase, int half) {
    Frag f;
    const unsigned short* p = p0 + col * stride + kbase + (half ? 8 : 0);
    f.q4[0] = *(const uint4*)(p);
    f.q4[1] = *(const uint4*)(p + 16);
    return f.v;
}

__global__ void prep_w_bf16(const float* __restrict__ w, unsigned short* __restrict__ wbf, int n) {
    int i = blockIdx.x * blockDim.x + threadIdx.x;
    if (i < n) wbf[i] = f2bf(w[i]);
}

// biasmat[h][qt][kt] = bias_table[rel_index(qt,kt)][h]
__global__ void prep_bias(const float* __restrict__ table, float* __restrict__ bm) {
    int i = blockIdx.x * blockDim.x + threadIdx.x;
    if (i >= HEADS * TOK * TOK) return;
    int kt = i & 63, qt = (i >> 6) & 63, h = i >> 12;
    int qy = qt >> 3, qx = qt & 7, ky = kt >> 3, kx = kt & 7;
    int idx = (qy - ky + WSZ - 1) * (2 * WSZ - 1) + (qx - kx + WSZ - 1);
    bm[i] = table[idx * HEADS + h];
}

__global__ __launch_bounds__(128)
void swin_attn_fused(const float* __restrict__ x,
                     const float* __restrict__ b_qkv,
                     const unsigned short* __restrict__ wbf,
                     const float* __restrict__ biasmat,
                     float* __restrict__ out) {
    extern __shared__ __align__(16) char smem[];
    // layout: xw [64][256] bf16 (32 KB), then per-wave 28 KB regions:
    //   qs [64][32] (4K) | ks [64][32] (4K) | vsT [32][64] (4K) |
    //   scratch 16K = { phase2: wsl [32][256] bf16 } { phase3: ps [64][64] bf16 (8K) }
    //                 { phase4: osT [32][64] f32 at +8K }
    unsigned short* xw = (unsigned short*)smem;

    const int lane = threadIdx.x & 31;
    const int wave = threadIdx.x >> 5;         // 0..3
    const int half = lane >> 4;
    const int l15  = lane & 15;

    int bid = blockIdx.x;
    const int hb = bid & 1;  bid >>= 1;        // head half: heads [hb*4, hb*4+4)
    const int wx = bid & (NWIN - 1); bid >>= 5;
    const int wy = bid & (NWIN - 1); bid >>= 5;
    const int b  = bid;
    const int head = hb * 4 + wave;

    const unsigned waveOff = 32u * 1024 + (unsigned)wave * WAVE_REGION;
    unsigned short* qs  = (unsigned short*)(smem + waveOff);
    unsigned short* ks  = (unsigned short*)(smem + waveOff + 4 * 1024);
    unsigned short* vsT = (unsigned short*)(smem + waveOff + 8 * 1024);
    unsigned short* wsl = (unsigned short*)(smem + waveOff + 12 * 1024);
    unsigned short* ps  = (unsigned short*)(smem + waveOff + 12 * 1024);
    float*          osT = (float*)         (smem + waveOff + 20 * 1024);
    const unsigned  wslOff = waveOff + 12u * 1024;

    // ---- async copy of one 16 KB W slice (32 rows x 256 ch bf16, contiguous) ----
    auto async_copy_wslice = [&](int s) {
        const unsigned short* gsrc = wbf + (size_t)(s * CH + head * HC) * CH;
        unsigned ldsOff = wslOff + (unsigned)lane * 16u;
        unsigned gOff   = (unsigned)lane * 16u;
        for (int i = 0; i < 32; ++i) {          // 32 x (32 lanes x 16 B) = 16 KB
            asm volatile("global_load_async_to_lds_b128 %0, %1, %2"
                         :: "v"(ldsOff), "v"(gOff), "s"(gsrc) : "memory");
            ldsOff += 512u; gOff += 512u;
        }
    };

    // kick off first W-slice copy; it overlaps the x-window staging below
    async_copy_wslice(0);

    // ---------- 1) cooperative stage of shifted x window -> LDS bf16 ----------
    const int x0 = wx * WSZ + WSZ / 2;          // shifted column base (4..252)
    const bool wrapx = (x0 + WSZ > HW);
    {
        const int cg = threadIdx.x & 31;        // channel group: 8 consecutive channels
        const int rg = threadIdx.x >> 5;        // row group: 2 rows
        const int c0 = cg * 8;
        for (int rr = 0; rr < 2; ++rr) {
            const int r = rg * 2 + rr;
            const int y = (wy * WSZ + r + WSZ / 2) & (HW - 1);
            float f[8][8];
            if (!wrapx) {
#pragma unroll
                for (int ch = 0; ch < 8; ++ch) {
                    const float* rowp = x + ((size_t)(b * CH + c0 + ch)) * (HW * HW) + y * HW;
                    *(float4*)&f[ch][0] = *(const float4*)(rowp + x0);
                    *(float4*)&f[ch][4] = *(const float4*)(rowp + x0 + 4);
                }
            } else {
#pragma unroll
                for (int ch = 0; ch < 8; ++ch) {
                    const float* rowp = x + ((size_t)(b * CH + c0 + ch)) * (HW * HW) + y * HW;
                    for (int cc = 0; cc < 8; ++cc) f[ch][cc] = rowp[(x0 + cc) & (HW - 1)];
                }
            }
#pragma unroll
            for (int cc = 0; cc < 8; ++cc) {    // pack 8 channels -> one b128 store
                uint4 w;
                w.x = pack2bf(f[0][cc], f[1][cc]);
                w.y = pack2bf(f[2][cc], f[3][cc]);
                w.z = pack2bf(f[4][cc], f[5][cc]);
                w.w = pack2bf(f[6][cc], f[7][cc]);
                *(uint4*)&xw[(r * 8 + cc) * CH + c0] = w;
            }
        }
    }
    __syncthreads();

    // ---------- 2) QKV GEMM: 64 tok x 32 outch x 256 inch per section ----------
    for (int s = 0; s < 3; ++s) {
        asm volatile("s_wait_asynccnt 0x0" ::: "memory");   // W slice resident
        const int wrow0 = s * CH + head * HC;
        for (int m = 0; m < 4; ++m) {
            for (int n = 0; n < 2; ++n) {
                v8f acc = {0.f, 0.f, 0.f, 0.f, 0.f, 0.f, 0.f, 0.f};
#pragma unroll
                for (int kk = 0; kk < 8; ++kk) {
                    v16bf a  = ld_afrag(xw, CH, m * 16 + l15, kk * 32, half);
                    v16bf bb = ld_bfrag(wsl, CH, n * 16 + l15, kk * 32, half);
                    acc = wmma_bf16(a, bb, acc);
                }
                const float badd = b_qkv[wrow0 + n * 16 + l15];
                if (s == 2) {                   // v stored transposed, packed b128
                    unsigned short tmp[8];
#pragma unroll
                    for (int r = 0; r < 8; ++r) tmp[r] = f2bf(acc[r] + badd);
                    *(uint4*)&vsT[(n * 16 + l15) * TOK + m * 16 + 8 * half] = *(uint4*)tmp;
                } else {
                    unsigned short* dst = (s == 0) ? qs : ks;
#pragma unroll
                    for (int r = 0; r < 8; ++r)
                        dst[(m * 16 + r + 8 * half) * HC + n * 16 + l15] = f2bf(acc[r] + badd);
                }
            }
        }
        if (s < 2) {
            asm volatile("s_wait_dscnt 0x0" ::: "memory");  // wsl reads drained
            async_copy_wslice(s + 1);                       // overlaps stores above
        }
    }

    // ---------- 3) S = q k^T * scale + bias, row softmax per 16-row strip ----------
    const float scale = 0.17677669529663687f;   // 32^-0.5
    const float* bm = biasmat + head * (TOK * TOK);
    for (int mt = 0; mt < 4; ++mt) {
        v8f Sacc[4];
        v16bf a = ld_afrag(qs, HC, mt * 16 + l15, 0, half);  // K=32 in one shot
#pragma unroll
        for (int nt = 0; nt < 4; ++nt) {
            v16bf bb = ld_bfrag(ks, HC, nt * 16 + l15, 0, half);
            v8f z = {0.f, 0.f, 0.f, 0.f, 0.f, 0.f, 0.f, 0.f};
            Sacc[nt] = wmma_bf16(a, bb, z);
        }
#pragma unroll
        for (int nt = 0; nt < 4; ++nt)
#pragma unroll
            for (int r = 0; r < 8; ++r) {
                const int qt = mt * 16 + r + 8 * half;
                Sacc[nt][r] = Sacc[nt][r] * scale + bm[qt * TOK + nt * 16 + l15];
            }
#pragma unroll
        for (int r = 0; r < 8; ++r) {
            float mx = fmaxf(fmaxf(Sacc[0][r], Sacc[1][r]), fmaxf(Sacc[2][r], Sacc[3][r]));
            for (int d = 1; d < 16; d <<= 1) mx = fmaxf(mx, __shfl_xor(mx, d, 16));
            float e0 = __expf(Sacc[0][r] - mx), e1 = __expf(Sacc[1][r] - mx);
            float e2 = __expf(Sacc[2][r] - mx), e3 = __expf(Sacc[3][r] - mx);
            float sum = e0 + e1 + e2 + e3;
            for (int d = 1; d < 16; d <<= 1) sum += __shfl_xor(sum, d, 16);
            const float inv = 1.0f / sum;
            const int qt = mt * 16 + r + 8 * half;
            ps[qt * TOK +  0 + l15] = f2bf(e0 * inv);
            ps[qt * TOK + 16 + l15] = f2bf(e1 * inv);
            ps[qt * TOK + 32 + l15] = f2bf(e2 * inv);
            ps[qt * TOK + 48 + l15] = f2bf(e3 * inv);
        }
    }

    // ---------- 4) O = P v ----------
    v8f Oacc[4][2];
#pragma unroll
    for (int mt = 0; mt < 4; ++mt)
#pragma unroll
        for (int n = 0; n < 2; ++n)
            Oacc[mt][n] = (v8f){0.f, 0.f, 0.f, 0.f, 0.f, 0.f, 0.f, 0.f};

    for (int mt = 0; mt < 4; ++mt)
#pragma unroll
        for (int kk = 0; kk < 2; ++kk) {
            v16bf a = ld_afrag(ps, TOK, mt * 16 + l15, kk * 32, half);
#pragma unroll
            for (int n = 0; n < 2; ++n) {
                v16bf bb = ld_bfrag(vsT, TOK, n * 16 + l15, kk * 32, half);
                Oacc[mt][n] = wmma_bf16(a, bb, Oacc[mt][n]);
            }
        }
    // O stored transposed ([hc][token]) so D-layout rows become contiguous b128s
#pragma unroll
    for (int mt = 0; mt < 4; ++mt)
#pragma unroll
        for (int n = 0; n < 2; ++n) {
            float* dst = osT + (n * 16 + l15) * TOK + mt * 16 + 8 * half;
            float4 lo, hi;
            lo.x = Oacc[mt][n][0]; lo.y = Oacc[mt][n][1];
            lo.z = Oacc[mt][n][2]; lo.w = Oacc[mt][n][3];
            hi.x = Oacc[mt][n][4]; hi.y = Oacc[mt][n][5];
            hi.z = Oacc[mt][n][6]; hi.w = Oacc[mt][n][7];
            *(float4*)dst = lo; *(float4*)(dst + 4) = hi;
        }

    // ---------- 5) coalesced store with inverse roll ----------
    for (int i = 0; i < 8; ++i) {
        const int id = lane + i * 32;           // 0..255 -> (hc, window row)
        const int c  = id & 31;
        const int r  = id >> 5;
        const int y  = (wy * WSZ + r + WSZ / 2) & (HW - 1);
        const float* sp = osT + c * TOK + r * 8;
        float4 f0 = *(const float4*)sp;
        float4 f1 = *(const float4*)(sp + 4);
        float* ob = out + (((size_t)(b * CH + head * HC + c)) * HW + y) * HW;
        if (!wrapx) {
            *(float4*)(ob + x0)     = f0;
            *(float4*)(ob + x0 + 4) = f1;
        } else {
            ob[(x0 + 0) & (HW - 1)] = f0.x; ob[(x0 + 1) & (HW - 1)] = f0.y;
            ob[(x0 + 2) & (HW - 1)] = f0.z; ob[(x0 + 3) & (HW - 1)] = f0.w;
            ob[(x0 + 4) & (HW - 1)] = f1.x; ob[(x0 + 5) & (HW - 1)] = f1.y;
            ob[(x0 + 6) & (HW - 1)] = f1.z; ob[(x0 + 7) & (HW - 1)] = f1.w;
        }
    }
}

extern "C" void kernel_launch(void* const* d_in, const int* in_sizes, int n_in,
                              void* d_out, int out_size, void* d_ws, size_t ws_size,
                              hipStream_t stream) {
    const float* x          = (const float*)d_in[0];
    const float* w_qkv      = (const float*)d_in[1];
    const float* b_qkv      = (const float*)d_in[2];
    const float* bias_table = (const float*)d_in[3];
    float* out = (float*)d_out;

    const int B = in_sizes[0] / (CH * HW * HW);   // 2

    unsigned short* wbf = (unsigned short*)d_ws;                        // 768*256 bf16
    float* biasmat = (float*)((char*)d_ws + (size_t)(3 * CH) * CH * 2); // 8*64*64 f32

    prep_w_bf16<<<(3 * CH * CH + 255) / 256, 256, 0, stream>>>(w_qkv, wbf, 3 * CH * CH);
    prep_bias<<<(HEADS * TOK * TOK + 255) / 256, 256, 0, stream>>>(bias_table, biasmat);

    (void)hipFuncSetAttribute((const void*)swin_attn_fused,
                              hipFuncAttributeMaxDynamicSharedMemorySize, LDS_TOTAL);

    dim3 grid(B * NWIN * NWIN * 2);               // 2 head-halves per window
    swin_attn_fused<<<grid, 128, LDS_TOTAL, stream>>>(x, b_qkv, wbf, biasmat, out);
}
